// NeRFPP_19456201851145
// MI455X (gfx1250) — matrix operand order repaired and checked by reading, not compiled, not run on hardware
//
#include <hip/hip_runtime.h>
#include <hip/hip_bf16.h>

// ---------------------------------------------------------------------------
// NeRF++ fused forward for gfx1250 (MI455X).
// One block = 2 rays x 64 samples = 128 points. 256 threads = 8 wave32.
// GEMMs via v_wmma_f32_16x16x32_f16. Weights are pre-converted once into d_ws
// as f16 in the exact padded LDS chunk layout, then streamed into LDS with
// GLOBAL_LOAD_ASYNC_TO_LDS_B128 (double buffered, s_wait_asynccnt).
// ---------------------------------------------------------------------------

typedef __attribute__((ext_vector_type(16))) _Float16 v16h;
typedef __attribute__((ext_vector_type(8)))  _Float16 v8h;
typedef __attribute__((ext_vector_type(8)))  float    v8f;
typedef __attribute__((ext_vector_type(4)))  int      v4i;

#define ASTR 264   // activation row stride (halves), 256 + 8 pad
#define PSTR 104   // pts-embed row stride (halves), 96 + 8 pad
#define VSTR 40    // views row stride (halves), 32 + 8 pad
#define WSTR 40    // weight-chunk K stride (halves), 32 + 8 pad

#if defined(__gfx1250__) && __has_builtin(__builtin_amdgcn_global_load_async_to_lds_b128)
#define ASYNC_LDS 1
#else
#define ASYNC_LDS 0
#endif

// ---- pre-transposed weight arena layout (halves), all compile-time ---------
// 12 GEMM passes per net: L0, L1..L4, L5a(pts), L5b(h), L6, L7, Wf, Wv_a, Wv_b
constexpr int pass_off(int net, int p) {
    constexpr int in_[12]  = {0, 20480, 102400, 184320, 266240, 348160,
                              368640, 450560, 532480, 614400, 696320, 737280};
    constexpr int out_[12] = {0, 30720, 112640, 194560, 276480, 358400,
                              389120, 471040, 552960, 634880, 716800, 757760};
    return net ? out_[p] : in_[p];
}
constexpr int NET_OUT_BASE    = 742400;            // halves
constexpr int WT_TOTAL_HALVES = 742400 + 762880;   // 1,505,280
// f32 arena (per-net stride F32_PER_NET floats): b0..b7 | ba | bf | bv | br | Wa | Wr
constexpr int F32_PER_NET = 3080;
constexpr int FB_BA = 2048, FB_BF = 2049, FB_BV = 2305, FB_BR = 2433;
constexpr int FB_WA = 2436, FB_WR = 2692;

struct NetParams {
    const float* W[8];
    const float* b[8];
    const float *Wa, *ba, *Wf, *bf, *Wv, *bv, *Wr, *br;
};

struct KArgsFlt {           // fallback path (raw f32 weights)
    const float *points, *out_points, *viewdirs, *in_z, *in_zmax;
    NetParams pin, pout;
    float *out_rgb, *out_w;
};

struct KArgsPre {           // fast path (pre-transposed f16 weights in ws)
    const float *points, *out_points, *viewdirs, *in_z, *in_zmax;
    const _Float16* wt;     // f16 chunk arena
    const float* fb;        // f32 bias/head arena
    float *out_rgb, *out_w;
};

struct Ctx {
    _Float16 *actA, *actB, *pts, *vbuf, *wc0, *wc1;
    int wave, lane;
};

// ---------------------------------------------------------------------------
// Weight prep kernel: f32 row-major -> f16, chunked [chunk][col][k] with
// WSTR padding and zero-fill of padded K rows; plus f32 bias/head copies.
// ---------------------------------------------------------------------------
struct ConvPass { const float* src; int row0, kreal, kchunks, nout, dstoff; };
struct F32Copy  { const float* src; int n, dstoff; };
struct ConvArgs {
    ConvPass p[24];
    F32Copy  f[28];
    _Float16* wt;
    float*    fb;
};

__global__ void convert_weights(ConvArgs ca) {
    int b = blockIdx.x;
    if (b < 24) {
        ConvPass d = ca.p[b];
        int per_chunk = d.nout * WSTR;
        int total = d.kchunks * per_chunk;
        _Float16* dst = ca.wt + d.dstoff;
        for (int idx = threadIdx.x; idx < total; idx += 256) {
            int chunk = idx / per_chunk;
            int rem   = idx - chunk * per_chunk;
            int col   = rem / WSTR;
            int k     = rem - col * WSTR;
            int kg    = chunk * 32 + k;
            float v = 0.f;
            if (k < 32 && kg < d.kreal)
                v = d.src[(size_t)(d.row0 + kg) * d.nout + col];
            dst[idx] = (_Float16)v;
        }
    } else {
        int base = (b - 24) * 14;
        for (int e = 0; e < 14; ++e) {
            F32Copy fc = ca.f[base + e];
            for (int idx = threadIdx.x; idx < fc.n; idx += 256)
                ca.fb[fc.dstoff + idx] = fc.src[idx];
        }
    }
}

// ---------------------------------------------------------------------------
// Async (or fallback) copy of one pre-transposed f16 chunk global -> LDS.
// Builtin signature (from clang diagnostic): arg0 = v4i addrspace(1)*,
// arg1 = LDS dest, then imm offset + imm cpol.
// ---------------------------------------------------------------------------
__device__ inline void copy_chunk(_Float16* lds, const _Float16* g, int halves) {
    int nb = halves >> 3;  // b128 blocks
#if ASYNC_LDS
    for (int i = threadIdx.x; i < nb; i += 256) {
        auto gp = (__attribute__((address_space(1))) v4i*)(void*)(g + i * 8);
        auto lp = (__attribute__((address_space(3))) v4i*)(void*)(lds + i * 8);
        __builtin_amdgcn_global_load_async_to_lds_b128(gp, lp, 0, 0);
    }
#else
    for (int i = threadIdx.x; i < nb; i += 256)
        *(v8h*)(lds + i * 8) = *(const v8h*)(g + i * 8);
#endif
}

__device__ inline void wait_async() {
#if ASYNC_LDS
#if __has_builtin(__builtin_amdgcn_s_wait_asynccnt)
    __builtin_amdgcn_s_wait_asynccnt(0);
#else
    asm volatile("s_wait_asynccnt 0" ::: "memory");
#endif
#endif
}

// ---------------------------------------------------------------------------
// WMMA fragment helpers (16-bit A 16x32 layout / B 32x16 layout / f32 C 16x16)
// ---------------------------------------------------------------------------
__device__ inline v16h load_a_frag(const _Float16* A, int astride, int kc,
                                   const Ctx& c) {
    int row = c.wave * 16 + (c.lane & 15);
    int kb  = kc * 32 + ((c.lane >> 4) << 3);
    const _Float16* ap = A + row * astride + kb;
    v8h lo = *(const v8h*)ap;
    v8h hi = *(const v8h*)(ap + 16);
    v16h r;
#pragma unroll
    for (int i = 0; i < 8; ++i) { r[i] = lo[i]; r[i + 8] = hi[i]; }
    return r;
}

template <int NT>
__device__ inline void wmma_tiles(v8f* C, const v16h& afr, const _Float16* cur,
                                  const Ctx& c) {
    int colb = c.lane & 15;
    int kb2  = (c.lane >> 4) << 3;
#pragma unroll
    for (int nt = 0; nt < NT; ++nt) {
        const _Float16* bp = cur + (nt * 16 + colb) * WSTR + kb2;
        v8h blo = *(const v8h*)bp;
        v8h bhi = *(const v8h*)(bp + 16);
        v16h bfr;
#pragma unroll
        for (int i = 0; i < 8; ++i) { bfr[i] = blo[i]; bfr[i + 8] = bhi[i]; }
        C[nt] = __builtin_amdgcn_wmma_f32_16x16x32_f16(
            false, afr, false, bfr, (short)0, C[nt], false, false);
    }
}

// ---- fast path: pre-transposed f16 weights streamed async ------------------
template <int NOUT>
__device__ inline void gemm_pass_pre(v8f* C, const _Float16* A, int astride,
                                     int kchunks, const _Float16* Wt,
                                     const Ctx& c) {
    constexpr int NT = NOUT / 16;
    constexpr int CH = NOUT * WSTR;
    copy_chunk(c.wc0, Wt, CH);
    wait_async();
    __syncthreads();
    for (int kc = 0; kc < kchunks; ++kc) {
        _Float16* cur = (kc & 1) ? c.wc1 : c.wc0;
        if (kc + 1 < kchunks)
            copy_chunk((kc & 1) ? c.wc0 : c.wc1, Wt + (kc + 1) * CH, CH);
        v16h afr = load_a_frag(A, astride, kc, c);
        wmma_tiles<NT>(C, afr, cur, c);
        wait_async();
        __syncthreads();
    }
}

// ---- fallback path: raw f32 weights, convert while staging -----------------
template <int NOUT>
__device__ inline void load_wchunk(_Float16* dst, const float* Wg,
                                   int row0, int rows_real) {
    constexpr int TOT = 32 * NOUT;
    for (int idx = threadIdx.x; idx < TOT; idx += 256) {
        int k   = idx / NOUT;
        int col = idx - k * NOUT;
        float v = (k < rows_real) ? Wg[(size_t)(row0 + k) * NOUT + col] : 0.0f;
        dst[col * WSTR + k] = (_Float16)v;
    }
}

template <int NOUT>
__device__ inline void gemm_pass_flt(v8f* C, const _Float16* A, int astride,
                                     int kchunks, int kreal,
                                     const float* Wg, int wrow0, const Ctx& c) {
    constexpr int NT = NOUT / 16;
    {
        int rr = kreal > 32 ? 32 : (kreal < 0 ? 0 : kreal);
        load_wchunk<NOUT>(c.wc0, Wg, wrow0, rr);
    }
    __syncthreads();
    for (int kc = 0; kc < kchunks; ++kc) {
        _Float16* cur = (kc & 1) ? c.wc1 : c.wc0;
        if (kc + 1 < kchunks) {
            int r0 = (kc + 1) * 32;
            int rr = kreal - r0;
            rr = rr < 0 ? 0 : (rr > 32 ? 32 : rr);
            load_wchunk<NOUT>((kc & 1) ? c.wc0 : c.wc1, Wg, wrow0 + r0, rr);
        }
        v16h afr = load_a_frag(A, astride, kc, c);
        wmma_tiles<NT>(C, afr, cur, c);
        __syncthreads();
    }
}

// ---- bias + relu + store to canonical activation layout --------------------
template <int NOUT>
__device__ inline void store_act(const v8f* C, const float* bias, bool relu,
                                 _Float16* dst, int dstride, const Ctx& c) {
    constexpr int NT = NOUT / 16;
    int colb   = c.lane & 15;
    int rowoff = c.wave * 16 + ((c.lane >> 4) << 3);
#pragma unroll
    for (int nt = 0; nt < NT; ++nt) {
        int col  = nt * 16 + colb;
        float bv = bias[col];
#pragma unroll
        for (int r = 0; r < 8; ++r) {
            float v = C[nt][r] + bv;
            if (relu) v = fmaxf(v, 0.f);
            dst[(rowoff + r) * dstride + col] = (_Float16)v;
        }
    }
    __syncthreads();
}

template <int NOUT>
__device__ inline void layerP(const Ctx& c, const _Float16* A, int astride,
                              int kchunks, const _Float16* Wt,
                              const float* bias, bool relu,
                              _Float16* dst, int dstride,
                              const _Float16* A2 = nullptr, int a2stride = 0,
                              int k2chunks = 0, const _Float16* Wt2 = nullptr) {
    constexpr int NT = NOUT / 16;
    v8f C[NT];
    v8f z = {0.f, 0.f, 0.f, 0.f, 0.f, 0.f, 0.f, 0.f};
#pragma unroll
    for (int nt = 0; nt < NT; ++nt) C[nt] = z;
    gemm_pass_pre<NOUT>(C, A, astride, kchunks, Wt, c);
    if (A2) gemm_pass_pre<NOUT>(C, A2, a2stride, k2chunks, Wt2, c);
    store_act<NOUT>(C, bias, relu, dst, dstride, c);
}

template <int NOUT>
__device__ inline void layerF(const Ctx& c, const _Float16* A, int astride,
                              int kpad, int kreal, const float* Wg,
                              const float* bias, bool relu,
                              _Float16* dst, int dstride,
                              const _Float16* A2 = nullptr, int a2stride = 0,
                              int k2pad = 0, int k2real = 0, int w2row0 = 0) {
    constexpr int NT = NOUT / 16;
    v8f C[NT];
    v8f z = {0.f, 0.f, 0.f, 0.f, 0.f, 0.f, 0.f, 0.f};
#pragma unroll
    for (int nt = 0; nt < NT; ++nt) C[nt] = z;
    gemm_pass_flt<NOUT>(C, A, astride, kpad >> 5, kreal, Wg, 0, c);
    if (A2) gemm_pass_flt<NOUT>(C, A2, a2stride, k2pad >> 5, k2real, Wg, w2row0, c);
    store_act<NOUT>(C, bias, relu, dst, dstride, c);
}

// ---------------------------------------------------------------------------
// Embeddings
// ---------------------------------------------------------------------------
__device__ inline void fill_pts(const Ctx& c, const float* __restrict__ src,
                                int D0, int Dreal, int Dpad, int base) {
    int t = threadIdx.x;
    if (t < 128) {
        const float* p = src + (size_t)(base + t) * D0;
        _Float16* row = c.pts + t * PSTR;
        float comp[4];
        for (int j = 0; j < D0; ++j) { comp[j] = p[j]; row[j] = (_Float16)comp[j]; }
        float f = 1.f;
        for (int l = 0; l < 10; ++l) {
            int b = D0 + l * 2 * D0;
            for (int j = 0; j < D0; ++j) {
                float v = comp[j] * f;
                row[b + j]      = (_Float16)__sinf(v);
                row[b + D0 + j] = (_Float16)__cosf(v);
            }
            f *= 2.f;
        }
        for (int j = Dreal; j < Dpad; ++j) row[j] = (_Float16)0.f;
    }
    __syncthreads();
}

__device__ inline void fill_views(const Ctx& c, const float* __restrict__ vd,
                                  int rayBase) {
    int t = threadIdx.x;
    if (t < 128) {
        int ray = rayBase + (t >> 6);
        _Float16* row = c.vbuf + t * VSTR;
        float comp[3];
        for (int j = 0; j < 3; ++j) { comp[j] = vd[ray * 3 + j]; row[j] = (_Float16)comp[j]; }
        float f = 1.f;
        for (int l = 0; l < 4; ++l) {
            int b = 3 + l * 6;
            for (int j = 0; j < 3; ++j) {
                float v = comp[j] * f;
                row[b + j]     = (_Float16)__sinf(v);
                row[b + 3 + j] = (_Float16)__cosf(v);
            }
            f *= 2.f;
        }
        for (int j = 27; j < 32; ++j) row[j] = (_Float16)0.f;
    }
    __syncthreads();
}

// ---------------------------------------------------------------------------
// Heads (vectorized v8h LDS reads)
// ---------------------------------------------------------------------------
__device__ inline void alpha_head(const Ctx& c, const float* Wa, float ba0,
                                  float* alpha_st) {
    int t = threadIdx.x;
    if (t < 128) {
        const _Float16* hrow = c.actB + t * ASTR;
        float s = 0.f;
        for (int k = 0; k < 256; k += 8) {
            v8h hv = *(const v8h*)(hrow + k);
#pragma unroll
            for (int j = 0; j < 8; ++j) s += (float)hv[j] * Wa[k + j];
        }
        alpha_st[t] = s + ba0;
    }
    __syncthreads();
}

__device__ inline void rgb_head(const Ctx& c, const float* Wr, const float* br,
                                float* rgb_st) {
    int t = threadIdx.x;
    if (t < 128) {
        const _Float16* hrow = c.actB + t * ASTR;
        float s0 = 0.f, s1 = 0.f, s2 = 0.f;
        for (int k = 0; k < 128; k += 8) {
            v8h hv = *(const v8h*)(hrow + k);
#pragma unroll
            for (int j = 0; j < 8; ++j) {
                float h = (float)hv[j];
                s0 += h * Wr[(k + j) * 3 + 0];
                s1 += h * Wr[(k + j) * 3 + 1];
                s2 += h * Wr[(k + j) * 3 + 2];
            }
        }
        rgb_st[t * 3 + 0] = s0 + br[0];
        rgb_st[t * 3 + 1] = s1 + br[1];
        rgb_st[t * 3 + 2] = s2 + br[2];
    }
    __syncthreads();
}

// ---------------------------------------------------------------------------
// MLP drivers
// ---------------------------------------------------------------------------
template <int NET>   // 0 = in-net (Dreal 63, 2 chunks), 1 = out-net (84, 3)
__device__ void run_mlp_pre(const Ctx& c, const _Float16* wtAll,
                            const float* fbAll, float* alpha_st, float* rgb_st) {
    constexpr int KC0 = NET ? 3 : 2;
    const _Float16* wt = wtAll + (NET ? NET_OUT_BASE : 0);
    const float* fb = fbAll + NET * F32_PER_NET;
    layerP<256>(c, c.pts,  PSTR, KC0, wt + pass_off(NET, 0), fb + 0,    true, c.actA, ASTR);
    layerP<256>(c, c.actA, ASTR, 8,   wt + pass_off(NET, 1), fb + 256,  true, c.actB, ASTR);
    layerP<256>(c, c.actB, ASTR, 8,   wt + pass_off(NET, 2), fb + 512,  true, c.actA, ASTR);
    layerP<256>(c, c.actA, ASTR, 8,   wt + pass_off(NET, 3), fb + 768,  true, c.actB, ASTR);
    layerP<256>(c, c.actB, ASTR, 8,   wt + pass_off(NET, 4), fb + 1024, true, c.actA, ASTR);
    layerP<256>(c, c.pts,  PSTR, KC0, wt + pass_off(NET, 5), fb + 1280, true, c.actB, ASTR,
                c.actA, ASTR, 8, wt + pass_off(NET, 6));
    layerP<256>(c, c.actB, ASTR, 8,   wt + pass_off(NET, 7), fb + 1536, true, c.actA, ASTR);
    layerP<256>(c, c.actA, ASTR, 8,   wt + pass_off(NET, 8), fb + 1792, true, c.actB, ASTR);
    alpha_head(c, fb + FB_WA, fb[FB_BA], alpha_st);
    layerP<256>(c, c.actB, ASTR, 8,   wt + pass_off(NET, 9), fb + FB_BF, false, c.actA, ASTR);
    layerP<128>(c, c.actA, ASTR, 8,   wt + pass_off(NET, 10), fb + FB_BV, true, c.actB, ASTR,
                c.vbuf, VSTR, 1, wt + pass_off(NET, 11));
    rgb_head(c, fb + FB_WR, fb + FB_BR, rgb_st);
}

__device__ void run_mlp_flt(const Ctx& c, const NetParams& P, int Dreal, int Dpad,
                            float* alpha_st, float* rgb_st) {
    layerF<256>(c, c.pts,  PSTR, Dpad, Dreal, P.W[0], P.b[0], true, c.actA, ASTR);
    layerF<256>(c, c.actA, ASTR, 256, 256,    P.W[1], P.b[1], true, c.actB, ASTR);
    layerF<256>(c, c.actB, ASTR, 256, 256,    P.W[2], P.b[2], true, c.actA, ASTR);
    layerF<256>(c, c.actA, ASTR, 256, 256,    P.W[3], P.b[3], true, c.actB, ASTR);
    layerF<256>(c, c.actB, ASTR, 256, 256,    P.W[4], P.b[4], true, c.actA, ASTR);
    layerF<256>(c, c.pts,  PSTR, Dpad, Dreal, P.W[5], P.b[5], true, c.actB, ASTR,
                c.actA, ASTR, 256, 256, Dreal);
    layerF<256>(c, c.actB, ASTR, 256, 256,    P.W[6], P.b[6], true, c.actA, ASTR);
    layerF<256>(c, c.actA, ASTR, 256, 256,    P.W[7], P.b[7], true, c.actB, ASTR);
    alpha_head(c, P.Wa, P.ba[0], alpha_st);
    layerF<256>(c, c.actB, ASTR, 256, 256, P.Wf, P.bf, false, c.actA, ASTR);
    layerF<128>(c, c.actA, ASTR, 256, 256, P.Wv, P.bv, true, c.actB, ASTR,
                c.vbuf, VSTR, 32, 27, 256);
    rgb_head(c, P.Wr, P.br, rgb_st);
}

// ---------------------------------------------------------------------------
// Shared LDS carve-up + volume rendering
// ---------------------------------------------------------------------------
__device__ inline Ctx make_ctx(char* smem, float*& fst) {
    _Float16* hb = (_Float16*)smem;
    Ctx c;
    c.actA = hb;
    c.actB = hb + 128 * ASTR;
    c.pts  = hb + 2 * 128 * ASTR;
    c.vbuf = c.pts + 128 * PSTR;
    c.wc0  = c.vbuf + 128 * VSTR;
    c.wc1  = c.wc0 + 256 * WSTR;
    fst    = (float*)(c.wc1 + 256 * WSTR);
    c.wave = threadIdx.x >> 5;
    c.lane = threadIdx.x & 31;
    return c;
}

__device__ void composite(int rayBase, const float* in_z, const float* in_zmax,
                          const float* out_points, float* out_rgb, float* out_w,
                          const float* in_a, const float* in_r,
                          const float* out_a, const float* out_r) {
    int t = threadIdx.x;
    if (t >= 2) return;
    int ray = rayBase + t;
    float T = 1.f, ri0 = 0.f, ri1 = 0.f, ri2 = 0.f;
    for (int s = 0; s < 64; ++s) {
        float z  = in_z[ray * 64 + s];
        float zn = (s < 63) ? in_z[ray * 64 + s + 1] : in_zmax[ray];
        float d  = zn - z;
        float ar = in_a[t * 64 + s];  ar = ar > 0.f ? ar : 0.f;
        float al = 1.f - __expf(-ar * d);
        float w  = al * T;
        out_w[(size_t)ray * 128 + s] = w;
        const float* rr = &in_r[(t * 64 + s) * 3];
        ri0 += w / (1.f + __expf(-rr[0]));
        ri1 += w / (1.f + __expf(-rr[1]));
        ri2 += w / (1.f + __expf(-rr[2]));
        T *= (1.f - al + 1e-10f);
    }
    float lastT = T;
    T = 1.f;
    float ro0 = 0.f, ro1 = 0.f, ro2 = 0.f;
    for (int s = 0; s < 64; ++s) {
        float z = out_points[(size_t)(ray * 64 + s) * 4 + 3];
        float d = (s < 63) ? (z - out_points[(size_t)(ray * 64 + s + 1) * 4 + 3])
                           : 1e10f;
        float ar = out_a[t * 64 + s];  ar = ar > 0.f ? ar : 0.f;
        float al = 1.f - __expf(-ar * d);
        float w  = al * T;
        out_w[(size_t)ray * 128 + 64 + s] = w;
        const float* rr = &out_r[(t * 64 + s) * 3];
        ro0 += w / (1.f + __expf(-rr[0]));
        ro1 += w / (1.f + __expf(-rr[1]));
        ro2 += w / (1.f + __expf(-rr[2]));
        T *= (1.f - al + 1e-10f);
    }
    out_rgb[ray * 3 + 0] = ri0 + lastT * ro0;
    out_rgb[ray * 3 + 1] = ri1 + lastT * ro1;
    out_rgb[ray * 3 + 2] = ri2 + lastT * ro2;
}

// ---------------------------------------------------------------------------
// Kernels
// ---------------------------------------------------------------------------
__global__ __launch_bounds__(256) void nerfpp_fused_pre(KArgsPre a) {
    extern __shared__ char smem[];
    float* fst;
    Ctx c = make_ctx(smem, fst);
    float* in_a  = fst;        float* in_r  = fst + 128;
    float* out_a = fst + 512;  float* out_r = fst + 640;
    int rayBase = blockIdx.x * 2;
    int ptBase  = rayBase * 64;

    fill_views(c, a.viewdirs, rayBase);
    fill_pts(c, a.points, 3, 63, 64, ptBase);
    run_mlp_pre<0>(c, a.wt, a.fb, in_a, in_r);
    __syncthreads();
    fill_pts(c, a.out_points, 4, 84, 96, ptBase);
    run_mlp_pre<1>(c, a.wt, a.fb, out_a, out_r);
    __syncthreads();
    composite(rayBase, a.in_z, a.in_zmax, a.out_points, a.out_rgb, a.out_w,
              in_a, in_r, out_a, out_r);
}

__global__ __launch_bounds__(256) void nerfpp_fused_flt(KArgsFlt a) {
    extern __shared__ char smem[];
    float* fst;
    Ctx c = make_ctx(smem, fst);
    float* in_a  = fst;        float* in_r  = fst + 128;
    float* out_a = fst + 512;  float* out_r = fst + 640;
    int rayBase = blockIdx.x * 2;
    int ptBase  = rayBase * 64;

    fill_views(c, a.viewdirs, rayBase);
    fill_pts(c, a.points, 3, 63, 64, ptBase);
    run_mlp_flt(c, a.pin, 63, 64, in_a, in_r);
    __syncthreads();
    fill_pts(c, a.out_points, 4, 84, 96, ptBase);
    run_mlp_flt(c, a.pout, 84, 96, out_a, out_r);
    __syncthreads();
    composite(rayBase, a.in_z, a.in_zmax, a.out_points, a.out_rgb, a.out_w,
              in_a, in_r, out_a, out_r);
}

// ---------------------------------------------------------------------------
// Host launch
// ---------------------------------------------------------------------------
extern "C" void kernel_launch(void* const* d_in, const int* in_sizes, int n_in,
                              void* d_out, int out_size, void* d_ws, size_t ws_size,
                              hipStream_t stream) {
    (void)n_in; (void)out_size;
    NetParams pin, pout;
    auto fill = [&](NetParams& P, int base) {
        for (int i = 0; i < 8; ++i) {
            P.W[i] = (const float*)d_in[base + 2 * i];
            P.b[i] = (const float*)d_in[base + 2 * i + 1];
        }
        P.Wa = (const float*)d_in[base + 16]; P.ba = (const float*)d_in[base + 17];
        P.Wf = (const float*)d_in[base + 18]; P.bf = (const float*)d_in[base + 19];
        P.Wv = (const float*)d_in[base + 20]; P.bv = (const float*)d_in[base + 21];
        P.Wr = (const float*)d_in[base + 22]; P.br = (const float*)d_in[base + 23];
    };
    fill(pin, 5);
    fill(pout, 29);

    int N = in_sizes[3] / 64;   // rays
    float* out_rgb = (float*)d_out;
    float* out_w   = (float*)d_out + (size_t)N * 3;

    size_t smem =
        (size_t)(2 * 128 * ASTR + 128 * PSTR + 128 * VSTR + 2 * 256 * WSTR) * 2
        + 1024 * 4;             // ~217 KB (< 320 KB WGP LDS)

    size_t ws_need = (size_t)WT_TOTAL_HALVES * 2 + (size_t)2 * F32_PER_NET * 4;
    if (d_ws != nullptr && ws_size >= ws_need) {
        // ---- build conversion descriptors (host mirrors of pass_off) ----
        static const int offIn[12]  = {0, 20480, 102400, 184320, 266240, 348160,
                                       368640, 450560, 532480, 614400, 696320, 737280};
        static const int offOut[12] = {0, 30720, 112640, 194560, 276480, 358400,
                                       389120, 471040, 552960, 634880, 716800, 757760};
        ConvArgs ca;
        ca.wt = (_Float16*)d_ws;
        ca.fb = (float*)((char*)d_ws + (size_t)WT_TOTAL_HALVES * 2);
        const NetParams* nets[2] = {&pin, &pout};
        for (int n = 0; n < 2; ++n) {
            const NetParams& P = *nets[n];
            int dreal = n ? 84 : 63;
            int kc0   = n ? 3 : 2;
            int nbase = n ? NET_OUT_BASE : 0;
            const int* off = n ? offOut : offIn;
            const float* src[12] = {P.W[0], P.W[1], P.W[2], P.W[3], P.W[4],
                                    P.W[5], P.W[5], P.W[6], P.W[7], P.Wf, P.Wv, P.Wv};
            int kreal[12] = {dreal, 256, 256, 256, 256, dreal, 256, 256, 256, 256, 256, 27};
            int row0[12]  = {0, 0, 0, 0, 0, 0, dreal, 0, 0, 0, 0, 256};
            int kch[12]   = {kc0, 8, 8, 8, 8, kc0, 8, 8, 8, 8, 8, 1};
            for (int p = 0; p < 12; ++p) {
                int nout = (p >= 10) ? 128 : 256;
                ca.p[n * 12 + p] = ConvPass{src[p], row0[p], kreal[p], kch[p],
                                            nout, nbase + off[p]};
            }
            int fb0 = n * F32_PER_NET;
            int e = n * 14;
            for (int i = 0; i < 8; ++i)
                ca.f[e + i] = F32Copy{P.b[i], 256, fb0 + i * 256};
            ca.f[e + 8]  = F32Copy{P.ba, 1,   fb0 + FB_BA};
            ca.f[e + 9]  = F32Copy{P.bf, 256, fb0 + FB_BF};
            ca.f[e + 10] = F32Copy{P.bv, 128, fb0 + FB_BV};
            ca.f[e + 11] = F32Copy{P.br, 3,   fb0 + FB_BR};
            ca.f[e + 12] = F32Copy{P.Wa, 256, fb0 + FB_WA};
            ca.f[e + 13] = F32Copy{P.Wr, 384, fb0 + FB_WR};
        }
        convert_weights<<<dim3(26), dim3(256), 0, stream>>>(ca);

        KArgsPre a;
        a.points = (const float*)d_in[0]; a.out_points = (const float*)d_in[1];
        a.viewdirs = (const float*)d_in[2];
        a.in_z = (const float*)d_in[3];   a.in_zmax = (const float*)d_in[4];
        a.wt = (const _Float16*)ca.wt;    a.fb = (const float*)ca.fb;
        a.out_rgb = out_rgb;              a.out_w = out_w;
        (void)hipFuncSetAttribute((const void*)nerfpp_fused_pre,
                                  hipFuncAttributeMaxDynamicSharedMemorySize,
                                  (int)smem);
        nerfpp_fused_pre<<<dim3(N / 2), dim3(256), smem, stream>>>(a);
    } else {
        KArgsFlt a;
        a.points = (const float*)d_in[0]; a.out_points = (const float*)d_in[1];
        a.viewdirs = (const float*)d_in[2];
        a.in_z = (const float*)d_in[3];   a.in_zmax = (const float*)d_in[4];
        a.pin = pin; a.pout = pout;
        a.out_rgb = out_rgb;              a.out_w = out_w;
        (void)hipFuncSetAttribute((const void*)nerfpp_fused_flt,
                                  hipFuncAttributeMaxDynamicSharedMemorySize,
                                  (int)smem);
        nerfpp_fused_flt<<<dim3(N / 2), dim3(256), smem, stream>>>(a);
    }
}
